// ExpertLayer_90692529422682
// MI455X (gfx1250) — matrix operand order repaired and checked by reading, not compiled, not run on hardware
//
#include <hip/hip_runtime.h>

// Problem sizes (fixed by the reference).
#define T_TOKENS 8192     // B*S
#define H_DIM    2048
#define N_EXP    8
#define I_EXP_DIM 1024
#define I_SH_DIM  4096

typedef __attribute__((ext_vector_type(16))) __bf16 v16bf;
typedef __attribute__((ext_vector_type(8)))  __bf16 v8bf;
typedef __attribute__((ext_vector_type(2)))  __bf16 v2bf;
typedef __attribute__((ext_vector_type(8)))  float  v8f;
typedef __attribute__((ext_vector_type(4)))  float  v4f;
typedef __attribute__((ext_vector_type(4)))  int    v4i;

#define LDS_STRIDE 48   // bf16 elems per LDS tile row (32 data + 16 pad); 96B rows keep b128 ops 16B-aligned
#define TILE_ELEMS (64 * LDS_STRIDE)   // elements per tile buffer

// ---- CDNA5 feature probes (compile-safe) ----------------------------------
#if defined(__has_builtin)
# if __has_builtin(__builtin_amdgcn_global_load_async_to_lds_b128)
#  define HAVE_ASYNC_LDS 1
# endif
# if __has_builtin(__builtin_amdgcn_s_wait_asynccnt)
#  define HAVE_WAIT_ASYNC 1
# endif
# if __has_builtin(__builtin_amdgcn_cvt_pk_bf16_f32)
#  define HAVE_PK_BF16 1
# endif
#endif

__device__ __forceinline__ void wait_async_lds() {
#if defined(HAVE_ASYNC_LDS)
# if defined(HAVE_WAIT_ASYNC)
    __builtin_amdgcn_s_wait_asynccnt(0);
# else
    asm volatile("s_wait_asynccnt 0" ::: "memory");
# endif
#endif
}

// fp32 -> bf16 (hardware conversion path; RNE).
__device__ __forceinline__ __bf16 f2bf(float f) { return (__bf16)f; }

__device__ __forceinline__ v8bf pack8_bf16(v4f f0, v4f f1) {
    v8bf b;
#if defined(HAVE_PK_BF16)
    v2bf p0 = __builtin_amdgcn_cvt_pk_bf16_f32(f0[0], f0[1]);
    v2bf p1 = __builtin_amdgcn_cvt_pk_bf16_f32(f0[2], f0[3]);
    v2bf p2 = __builtin_amdgcn_cvt_pk_bf16_f32(f1[0], f1[1]);
    v2bf p3 = __builtin_amdgcn_cvt_pk_bf16_f32(f1[2], f1[3]);
    b[0] = p0[0]; b[1] = p0[1]; b[2] = p1[0]; b[3] = p1[1];
    b[4] = p2[0]; b[5] = p2[1]; b[6] = p3[0]; b[7] = p3[1];
#else
    b[0] = (__bf16)f0[0]; b[1] = (__bf16)f0[1]; b[2] = (__bf16)f0[2]; b[3] = (__bf16)f0[3];
    b[4] = (__bf16)f1[0]; b[5] = (__bf16)f1[1]; b[6] = (__bf16)f1[2]; b[7] = (__bf16)f1[3];
#endif
    return b;
}

// Stage one 64x32 bf16 tile (16 B per thread). Async global->LDS when available.
// Builtin signature (from hipcc diagnostic): param0 = v4i addrspace(1)*, so
// (global src, lds dst, imm offset, cpol).
__device__ __forceinline__ void stage_tile(__bf16* ldsTile, const __bf16* __restrict__ src,
                                           long srcStride, int rowBase, int kBase, int tid) {
    const int row = tid >> 2;
    const int col = (tid & 3) << 3;
    const __bf16* g = src + (long)(rowBase + row) * srcStride + kBase + col;
    __bf16* d = ldsTile + row * LDS_STRIDE + col;
#if defined(HAVE_ASYNC_LDS)
    __builtin_amdgcn_global_load_async_to_lds_b128(
        (__attribute__((address_space(1))) v4i*)g,
        (__attribute__((address_space(3))) v4i*)d, 0, 0);
#else
    *(v8bf*)d = *(const v8bf*)g;
#endif
}

// A fragment (16x32 bf16, MxK). Lanes 0-15 row m: K{0..7,16..23}; lanes 16-31: K{8..15,24..31}.
__device__ __forceinline__ v16bf frag_a(const __bf16* tile, int mBase, int lane) {
    const int m  = mBase + (lane & 15);
    const int hi = (lane >> 4) & 1;
    v8bf lo = *(const v8bf*)(tile + m * LDS_STRIDE + (hi ? 8 : 0));
    v8bf hg = *(const v8bf*)(tile + m * LDS_STRIDE + (hi ? 24 : 16));
    v16bf r;
#pragma unroll
    for (int i = 0; i < 8; ++i) { r[i] = lo[i]; r[i + 8] = hg[i]; }
    return r;
}

// B fragment (32x16 bf16, KxN). Lanes 0-15: col n, K{0..7,16..23}; lanes 16-31: col n, K{8..15,24..31}.
__device__ __forceinline__ v16bf frag_b(const __bf16* tile, int nBase, int lane) {
    const int n  = nBase + (lane & 15);
    const int k0 = ((lane >> 4) & 1) ? 8 : 0;
    v8bf lo = *(const v8bf*)(tile + n * LDS_STRIDE + k0);
    v8bf hg = *(const v8bf*)(tile + n * LDS_STRIDE + k0 + 16);
    v16bf r;
#pragma unroll
    for (int i = 0; i < 8; ++i) { r[i] = lo[i]; r[i + 8] = hg[i]; }
    return r;
}

// ---------------------------------------------------------------------------
// One-shot fp32 -> bf16 conversion (8 elements / thread, n multiple of 2048).
// ---------------------------------------------------------------------------
__global__ __launch_bounds__(256)
void cvt_f32_bf16_kernel(const float* __restrict__ src, __bf16* __restrict__ dst, long n) {
    const long i = ((long)blockIdx.x * 256 + threadIdx.x) * 8;
    if (i >= n) return;
    v4f f0 = *(const v4f*)(src + i);
    v4f f1 = *(const v4f*)(src + i + 4);
    *(v8bf*)(dst + i) = pack8_bf16(f0, f1);
}

// ---------------------------------------------------------------------------
// Router: one wave32 per token; 8 logits + shared sigmoid gate.
// comb[T,8] = softmax probs at top-2 (else 0); sgate[T] = sigmoid(x . segw).
// ---------------------------------------------------------------------------
__global__ __launch_bounds__(256)
void router_kernel(const float* __restrict__ X, const float* __restrict__ GW,
                   const float* __restrict__ SEGW, float* __restrict__ comb,
                   float* __restrict__ sgate) {
    const int wave = (blockIdx.x * blockDim.x + threadIdx.x) >> 5;
    const int lane = threadIdx.x & 31;
    const float* xr = X + (long)wave * H_DIM;

    float s[9];
#pragma unroll
    for (int e = 0; e < 9; ++e) s[e] = 0.f;
    for (int h = lane; h < H_DIM; h += 32) {
        const float xv = xr[h];
#pragma unroll
        for (int e = 0; e < 8; ++e) s[e] += xv * GW[e * H_DIM + h];
        s[8] += xv * SEGW[h];
    }
#pragma unroll
    for (int off = 16; off > 0; off >>= 1) {
#pragma unroll
        for (int e = 0; e < 9; ++e) s[e] += __shfl_xor(s[e], off, 32);
    }
    if (lane == 0) {
        float m = s[0];
#pragma unroll
        for (int e = 1; e < 8; ++e) m = fmaxf(m, s[e]);
        float p[8], sum = 0.f;
#pragma unroll
        for (int e = 0; e < 8; ++e) { p[e] = __expf(s[e] - m); sum += p[e]; }
        const float inv = 1.f / sum;
#pragma unroll
        for (int e = 0; e < 8; ++e) p[e] *= inv;
        int i1 = 0;
#pragma unroll
        for (int e = 1; e < 8; ++e) if (p[e] > p[i1]) i1 = e;
        int i2 = (i1 == 0) ? 1 : 0;
#pragma unroll
        for (int e = 0; e < 8; ++e) if (e != i1 && p[e] > p[i2]) i2 = e;
#pragma unroll
        for (int e = 0; e < 8; ++e)
            comb[(long)wave * 8 + e] = (e == i1) ? p[i1] : ((e == i2) ? p[i2] : 0.f);
        sgate[wave] = 1.f / (1.f + __expf(-s[8]));
    }
}

// ---------------------------------------------------------------------------
// Fused gate/up GEMM (all-bf16 inputs): act = bf16( silu(x@Wg^T) * (x@Wu^T) ).
// Block 256 = 8 waves (2x4); tile 64(M)x64(N); K chunk 32; double-buffered LDS
// with async global->LDS staging overlapping the WMMAs.
// ---------------------------------------------------------------------------
__global__ __launch_bounds__(256)
void moe_gateup_kernel(const __bf16* __restrict__ X, const __bf16* __restrict__ Wg,
                       const __bf16* __restrict__ Wu, __bf16* __restrict__ act,
                       int N, int Kd) {
    __shared__ __bf16 sA [2][TILE_ELEMS];
    __shared__ __bf16 sBg[2][TILE_ELEMS];
    __shared__ __bf16 sBu[2][TILE_ELEMS];

    const int tid    = threadIdx.x;
    const int lane   = tid & 31;
    const int waveId = tid >> 5;
    const int wm = waveId >> 2;      // 0..1
    const int wn = waveId & 3;       // 0..3
    const int rowBaseT = blockIdx.x * 64;
    const int rowBaseN = blockIdx.y * 64;

    v8f accg0 = {}; v8f accg1 = {}; v8f accu0 = {}; v8f accu1 = {};

    const int kIters = Kd >> 5;
    // Prologue: stage k=0 into buffer 0.
    stage_tile(sA [0], X,  Kd, rowBaseT, 0, tid);
    stage_tile(sBg[0], Wg, Kd, rowBaseN, 0, tid);
    stage_tile(sBu[0], Wu, Kd, rowBaseN, 0, tid);
    wait_async_lds();
    __syncthreads();

    int buf = 0;
    for (int kb = 0; kb < kIters; ++kb) {
        if (kb + 1 < kIters) {
            const int kNext = (kb + 1) << 5;
            stage_tile(sA [buf ^ 1], X,  Kd, rowBaseT, kNext, tid);
            stage_tile(sBg[buf ^ 1], Wg, Kd, rowBaseN, kNext, tid);
            stage_tile(sBu[buf ^ 1], Wu, Kd, rowBaseN, kNext, tid);
            __builtin_prefetch(X  + (long)(rowBaseT + (tid >> 2)) * Kd + kNext + 32, 0, 2);
            __builtin_prefetch(Wg + (long)(rowBaseN + (tid >> 2)) * Kd + kNext + 32, 0, 2);
        }
        v16bf a0 = frag_a(sA [buf], wm * 32,      lane);
        v16bf a1 = frag_a(sA [buf], wm * 32 + 16, lane);
        v16bf bg = frag_b(sBg[buf], wn * 16,      lane);
        v16bf bu = frag_b(sBu[buf], wn * 16,      lane);

        accg0 = __builtin_amdgcn_wmma_f32_16x16x32_bf16(false, a0, false, bg, (short)0, accg0, false, false);
        accg1 = __builtin_amdgcn_wmma_f32_16x16x32_bf16(false, a1, false, bg, (short)0, accg1, false, false);
        accu0 = __builtin_amdgcn_wmma_f32_16x16x32_bf16(false, a0, false, bu, (short)0, accu0, false, false);
        accu1 = __builtin_amdgcn_wmma_f32_16x16x32_bf16(false, a1, false, bu, (short)0, accu1, false, false);

        wait_async_lds();
        __syncthreads();
        buf ^= 1;
    }

    // C layout: VGPR r, lanes 0-15 -> (M=r, N=lane); lanes 16-31 -> (M=r+8, N=lane-16).
    const int hi  = (lane >> 4) & 1;
    const int col = rowBaseN + wn * 16 + (lane & 15);
#pragma unroll
    for (int r = 0; r < 8; ++r) {
        const int t0 = rowBaseT + wm * 32 + r + hi * 8;
        float g = accg0[r], u = accu0[r];
        act[(long)t0 * N + col] = f2bf((g / (1.f + __expf(-g))) * u);
        const int t1 = t0 + 16;
        g = accg1[r]; u = accu1[r];
        act[(long)t1 * N + col] = f2bf((g / (1.f + __expf(-g))) * u);
    }
}

// ---------------------------------------------------------------------------
// Down-projection GEMM with per-token scaling:
//   out[t, n] (+)= scale[t*scaleStride] * (act @ Wd^T)[t, n]
// ---------------------------------------------------------------------------
__global__ __launch_bounds__(256)
void moe_down_kernel(const __bf16* __restrict__ A, const __bf16* __restrict__ Wd,
                     float* __restrict__ out, const float* __restrict__ scale,
                     int scaleStride, int Kd, int accum) {
    __shared__ __bf16 sA[2][TILE_ELEMS];
    __shared__ __bf16 sB[2][TILE_ELEMS];

    const int tid    = threadIdx.x;
    const int lane   = tid & 31;
    const int waveId = tid >> 5;
    const int wm = waveId >> 2;
    const int wn = waveId & 3;
    const int rowBaseT = blockIdx.x * 64;
    const int rowBaseN = blockIdx.y * 64;

    v8f acc0 = {}; v8f acc1 = {};

    const int kIters = Kd >> 5;
    stage_tile(sA[0], A,  Kd, rowBaseT, 0, tid);
    stage_tile(sB[0], Wd, Kd, rowBaseN, 0, tid);
    wait_async_lds();
    __syncthreads();

    int buf = 0;
    for (int kb = 0; kb < kIters; ++kb) {
        if (kb + 1 < kIters) {
            const int kNext = (kb + 1) << 5;
            stage_tile(sA[buf ^ 1], A,  Kd, rowBaseT, kNext, tid);
            stage_tile(sB[buf ^ 1], Wd, Kd, rowBaseN, kNext, tid);
            __builtin_prefetch(A  + (long)(rowBaseT + (tid >> 2)) * Kd + kNext + 32, 0, 2);
            __builtin_prefetch(Wd + (long)(rowBaseN + (tid >> 2)) * Kd + kNext + 32, 0, 2);
        }
        v16bf a0 = frag_a(sA[buf], wm * 32,      lane);
        v16bf a1 = frag_a(sA[buf], wm * 32 + 16, lane);
        v16bf b  = frag_b(sB[buf], wn * 16,      lane);

        acc0 = __builtin_amdgcn_wmma_f32_16x16x32_bf16(false, a0, false, b, (short)0, acc0, false, false);
        acc1 = __builtin_amdgcn_wmma_f32_16x16x32_bf16(false, a1, false, b, (short)0, acc1, false, false);

        wait_async_lds();
        __syncthreads();
        buf ^= 1;
    }

    const int hi  = (lane >> 4) & 1;
    const int col = rowBaseN + wn * 16 + (lane & 15);
#pragma unroll
    for (int r = 0; r < 8; ++r) {
        const int t0 = rowBaseT + wm * 32 + r + hi * 8;
        const int t1 = t0 + 16;
        const float sc0 = scale[(long)t0 * scaleStride];
        const float sc1 = scale[(long)t1 * scaleStride];
        const long  i0  = (long)t0 * H_DIM + col;
        const long  i1  = (long)t1 * H_DIM + col;
        const float v0 = sc0 * acc0[r];
        const float v1 = sc1 * acc1[r];
        if (accum) { out[i0] += v0; out[i1] += v1; }
        else       { out[i0]  = v0; out[i1]  = v1; }
    }
}

// ---------------------------------------------------------------------------
// Host-side launch. Inputs in setup_inputs() order.
// ---------------------------------------------------------------------------
extern "C" void kernel_launch(void* const* d_in, const int* in_sizes, int n_in,
                              void* d_out, int out_size, void* d_ws, size_t ws_size,
                              hipStream_t stream) {
    const float* X    = (const float*)d_in[0];  // [T, H]
    const float* GW   = (const float*)d_in[1];  // [8, H]
    const float* EG   = (const float*)d_in[2];  // [8, I_EXP, H]
    const float* EU   = (const float*)d_in[3];  // [8, I_EXP, H]
    const float* ED   = (const float*)d_in[4];  // [8, H, I_EXP]
    const float* SG   = (const float*)d_in[5];  // [I_SH, H]
    const float* SU   = (const float*)d_in[6];  // [I_SH, H]
    const float* SD   = (const float*)d_in[7];  // [H, I_SH]
    const float* SEGW = (const float*)d_in[8];  // [1, H]
    float* out = (float*)d_out;                 // [T, H]

    const long nX  = (long)T_TOKENS * H_DIM;          // 16.8M
    const long nE  = (long)N_EXP * I_EXP_DIM * H_DIM; // 16.8M (per expert tensor set)
    const long nSg = (long)I_SH_DIM * H_DIM;          // 8.4M
    const long nSd = (long)H_DIM * I_SH_DIM;          // 8.4M

    char* ws = (char*)d_ws;
    float*  comb  = (float*)ws;                          // T*8 floats
    float*  sgate = comb + (size_t)T_TOKENS * 8;         // T floats
    size_t  off   = (size_t)1 << 20;
    __bf16* xb  = (__bf16*)(ws + off); off += (size_t)nX  * 2;
    __bf16* egb = (__bf16*)(ws + off); off += (size_t)nE  * 2;
    __bf16* eub = (__bf16*)(ws + off); off += (size_t)nE  * 2;
    __bf16* edb = (__bf16*)(ws + off); off += (size_t)nE  * 2;
    __bf16* sgb = (__bf16*)(ws + off); off += (size_t)nSg * 2;
    __bf16* sub = (__bf16*)(ws + off); off += (size_t)nSg * 2;
    __bf16* sdb = (__bf16*)(ws + off); off += (size_t)nSd * 2;
    __bf16* act = (__bf16*)(ws + off);                   // T * I_SH bf16 (64 MB)

    const dim3 blk(256);

    // 0) One-shot fp32 -> bf16 conversions (weights converted once, not per tile).
    cvt_f32_bf16_kernel<<<nX  / 2048, blk, 0, stream>>>(X,  xb,  nX);
    cvt_f32_bf16_kernel<<<nE  / 2048, blk, 0, stream>>>(EG, egb, nE);
    cvt_f32_bf16_kernel<<<nE  / 2048, blk, 0, stream>>>(EU, eub, nE);
    cvt_f32_bf16_kernel<<<nE  / 2048, blk, 0, stream>>>(ED, edb, nE);
    cvt_f32_bf16_kernel<<<nSg / 2048, blk, 0, stream>>>(SG, sgb, nSg);
    cvt_f32_bf16_kernel<<<nSg / 2048, blk, 0, stream>>>(SU, sub, nSg);
    cvt_f32_bf16_kernel<<<nSd / 2048, blk, 0, stream>>>(SD, sdb, nSd);

    // 1) Router (fp32 precision, tiny).
    router_kernel<<<T_TOKENS / 8, blk, 0, stream>>>(X, GW, SEGW, comb, sgate);

    // 2) Shared expert: gate/up then sigmoid-scaled down-proj (initializes d_out).
    moe_gateup_kernel<<<dim3(T_TOKENS / 64, I_SH_DIM / 64), blk, 0, stream>>>(
        xb, sgb, sub, act, I_SH_DIM, H_DIM);
    moe_down_kernel<<<dim3(T_TOKENS / 64, H_DIM / 64), blk, 0, stream>>>(
        act, sdb, out, sgate, 1, I_SH_DIM, /*accum=*/0);

    // 3) Routed experts, dense (comb is zero for unselected), accumulating into d_out.
    for (int e = 0; e < N_EXP; ++e) {
        const __bf16* wg = egb + (size_t)e * I_EXP_DIM * H_DIM;
        const __bf16* wu = eub + (size_t)e * I_EXP_DIM * H_DIM;
        const __bf16* wd = edb + (size_t)e * H_DIM * I_EXP_DIM;
        moe_gateup_kernel<<<dim3(T_TOKENS / 64, I_EXP_DIM / 64), blk, 0, stream>>>(
            xb, wg, wu, act, I_EXP_DIM, H_DIM);
        moe_down_kernel<<<dim3(T_TOKENS / 64, H_DIM / 64), blk, 0, stream>>>(
            act, wd, out, comb + e, 8, I_EXP_DIM, /*accum=*/1);
    }
}
